// CrossCoder_40845138985222
// MI455X (gfx1250) — matrix operand order repaired and checked by reading, not compiled, not run on hardware
//
#include <hip/hip_runtime.h>

// Problem constants (from the reference)
#define B_   2048
#define H_   16384
#define MD_  4096   // NM * D = 2 * 2048

// GEMM tiling
#define BM   128
#define BN   128
#define BK   32
#define SAs  40     // LDS row stride (halfs) for A tiles (pad: conflict-free b128 reads)
#define SBs  40     // LDS row stride (halfs) for B tiles (stored transposed [n][k])

typedef __bf16 v16bf __attribute__((ext_vector_type(16)));
typedef float  v8f   __attribute__((ext_vector_type(8)));

union Frag { uint4 q[2]; v16bf v; };

// Order-preserving float -> uint key (larger float => larger key)
__device__ __forceinline__ unsigned f2key(float f) {
  unsigned u = __float_as_uint(f);
  return (u & 0x80000000u) ? ~u : (u | 0x80000000u);
}

__device__ __forceinline__ unsigned short f2bf_rn(float x) {
  unsigned u = __float_as_uint(x);
  unsigned r = u + 0x7FFFu + ((u >> 16) & 1u);   // round-to-nearest-even
  return (unsigned short)(r >> 16);
}

__device__ __forceinline__ void split_bf(float x, unsigned short& h, unsigned short& l) {
  h = f2bf_rn(x);
  float hf = __uint_as_float(((unsigned)h) << 16);
  l = f2bf_rn(x - hf);
}

// ---- tile global loads into registers (prefetch) ----
__device__ __forceinline__ void load_tileA(float4* pA, const float* __restrict__ Ap,
                                           int bm0, int K, int k0, int tid) {
  #pragma unroll
  for (int it = 0; it < 4; ++it) {
    int idx = tid + it * 256;
    int r   = idx >> 3;                  // row 0..127
    int c4  = (idx & 7) << 2;            // k base 0..28
    pA[it] = *(const float4*)(Ap + (size_t)(bm0 + r) * K + (k0 + c4));
  }
}

__device__ __forceinline__ void load_tileB(float4* pB, const float* __restrict__ Bp,
                                           int bn0, int N, int k0, int tid) {
  #pragma unroll
  for (int it = 0; it < 4; ++it) {
    int idx = tid + it * 256;
    int r   = idx >> 5;                  // k row 0..31
    int c4  = (idx & 31) << 2;           // n base 0..124
    pB[it] = *(const float4*)(Bp + (size_t)(k0 + r) * N + (bn0 + c4));
  }
}

// ---- convert + store register tiles to LDS (hi/lo bf16 split) ----
template<int MODE>
__device__ __forceinline__ void store_tileA(const float4* pA,
                                            unsigned short* sAhi, unsigned short* sAlo,
                                            int tid, int k0,
                                            const float* __restrict__ benc, unsigned thr) {
  #pragma unroll
  for (int it = 0; it < 4; ++it) {
    int idx = tid + it * 256;
    int r   = idx >> 3;
    int c4  = (idx & 7) << 2;
    float4 v = pA[it];
    if (MODE == 1) {
      int hb = k0 + c4;
      v.x = ((f2key(v.x) >= thr) ? v.x : 0.0f) + benc[hb + 0];
      v.y = ((f2key(v.y) >= thr) ? v.y : 0.0f) + benc[hb + 1];
      v.z = ((f2key(v.z) >= thr) ? v.z : 0.0f) + benc[hb + 2];
      v.w = ((f2key(v.w) >= thr) ? v.w : 0.0f) + benc[hb + 3];
    }
    unsigned short h0,h1,h2,h3,l0,l1,l2,l3;
    split_bf(v.x, h0, l0); split_bf(v.y, h1, l1);
    split_bf(v.z, h2, l2); split_bf(v.w, h3, l3);
    int o = r * SAs + c4;
    *(unsigned*)&sAhi[o]     = (unsigned)h0 | ((unsigned)h1 << 16);
    *(unsigned*)&sAhi[o + 2] = (unsigned)h2 | ((unsigned)h3 << 16);
    *(unsigned*)&sAlo[o]     = (unsigned)l0 | ((unsigned)l1 << 16);
    *(unsigned*)&sAlo[o + 2] = (unsigned)l2 | ((unsigned)l3 << 16);
  }
}

__device__ __forceinline__ void store_tileB(const float4* pB,
                                            unsigned short* sBhi, unsigned short* sBlo,
                                            int tid) {
  #pragma unroll
  for (int it = 0; it < 4; ++it) {
    int idx = tid + it * 256;
    int r   = idx >> 5;
    int c4  = (idx & 31) << 2;
    const float4 v = pB[it];
    unsigned short h, l;
    split_bf(v.x, h, l); sBhi[(c4+0)*SBs + r] = h; sBlo[(c4+0)*SBs + r] = l;
    split_bf(v.y, h, l); sBhi[(c4+1)*SBs + r] = h; sBlo[(c4+1)*SBs + r] = l;
    split_bf(v.z, h, l); sBhi[(c4+2)*SBs + r] = h; sBlo[(c4+2)*SBs + r] = l;
    split_bf(v.w, h, l); sBhi[(c4+3)*SBs + r] = h; sBlo[(c4+3)*SBs + r] = l;
  }
}

// ---------------------------------------------------------------------------
// bf16x3-split WMMA GEMM:  C(MxN) = A(MxK) @ B(KxN)
// MODE 0: encode  (A = x raw, C = x_enc)
// MODE 1: decode  (A = topk-mask(x_enc) + b_enc applied on the fly, C += b_dec)
// Register-prefetch pipelined: global loads for stage k+1 issue before the
// WMMA block of stage k, hiding HBM latency under the matrix pipe.
// ---------------------------------------------------------------------------
template<int MODE>
__global__ __launch_bounds__(256) void wmma_gemm_kernel(
    const float* __restrict__ Ap, const float* __restrict__ Bp,
    float* __restrict__ Cp, int M, int N, int K,
    const float* __restrict__ benc, const unsigned* __restrict__ sel,
    const float* __restrict__ bdec)
{
  __shared__ __align__(16) unsigned short sAhi[BM * SAs];
  __shared__ __align__(16) unsigned short sAlo[BM * SAs];
  __shared__ __align__(16) unsigned short sBhi[BN * SBs];
  __shared__ __align__(16) unsigned short sBlo[BN * SBs];

  const int tid  = threadIdx.x;
  const int lane = tid & 31;          // wave32
  const int wid  = tid >> 5;          // 8 waves
  const int half = lane >> 4;         // lane group (0: lanes 0-15, 1: lanes 16-31)
  const int lm   = lane & 15;

  const int bm0 = blockIdx.y * BM;
  const int bn0 = blockIdx.x * BN;

  const int wm = (wid & 3) * 32;      // 4 waves tile M
  const int wn = (wid >> 2) * 64;     // 2 waves tile N

  const unsigned thr = (MODE == 1) ? sel[4] : 0u;

  v8f acc[2][4];
  #pragma unroll
  for (int i = 0; i < 2; ++i)
    #pragma unroll
    for (int j = 0; j < 4; ++j)
      acc[i][j] = (v8f){0.f,0.f,0.f,0.f,0.f,0.f,0.f,0.f};

  float4 pA[4], pB[4];
  load_tileA(pA, Ap, bm0, K, 0, tid);
  load_tileB(pB, Bp, bn0, N, 0, tid);

  for (int k0 = 0; k0 < K; k0 += BK) {
    // fill LDS from prefetch registers (convert f32 -> hi/lo bf16)
    store_tileA<MODE>(pA, sAhi, sAlo, tid, k0, benc, thr);
    store_tileB(pB, sBhi, sBlo, tid);
    __syncthreads();

    // issue next stage's global loads early: latency hides under WMMA below
    if (k0 + BK < K) {
      load_tileA(pA, Ap, bm0, K, k0 + BK, tid);
      load_tileB(pB, Bp, bn0, N, k0 + BK, tid);
    }

    // gather fragments per ISA 16-bit layouts
    // A 16x32: lane(m=lm): dwords 0-3 => k = half*8..+7 ; dwords 4-7 => k = 16+half*8..
    Frag ah[2], al[2], bh[4], bl[4];
    #pragma unroll
    for (int i = 0; i < 2; ++i) {
      int row = wm + i * 16 + lm;
      const unsigned short* p = &sAhi[row * SAs + half * 8];
      ah[i].q[0] = *(const uint4*)p;
      ah[i].q[1] = *(const uint4*)(p + 16);
      const unsigned short* q = &sAlo[row * SAs + half * 8];
      al[i].q[0] = *(const uint4*)q;
      al[i].q[1] = *(const uint4*)(q + 16);
    }
    // B 32x16: lane(n=lm): dwords 0-3 => k = half*8..+7 ; dwords 4-7 => k = 16+half*8.. (matches A)
    #pragma unroll
    for (int j = 0; j < 4; ++j) {
      int col = wn + j * 16 + lm;
      const unsigned short* p = &sBhi[col * SBs + half * 8];
      bh[j].q[0] = *(const uint4*)p;
      bh[j].q[1] = *(const uint4*)(p + 16);
      const unsigned short* q = &sBlo[col * SBs + half * 8];
      bl[j].q[0] = *(const uint4*)q;
      bl[j].q[1] = *(const uint4*)(q + 16);
    }

    // bf16x3 split accumulation: hi*hi + hi*lo + lo*hi (f32 accumulate)
    #pragma unroll
    for (int i = 0; i < 2; ++i)
      #pragma unroll
      for (int j = 0; j < 4; ++j) {
        acc[i][j] = __builtin_amdgcn_wmma_f32_16x16x32_bf16(
            false, ah[i].v, false, bh[j].v, (short)0, acc[i][j], false, false);
        acc[i][j] = __builtin_amdgcn_wmma_f32_16x16x32_bf16(
            false, ah[i].v, false, bl[j].v, (short)0, acc[i][j], false, false);
        acc[i][j] = __builtin_amdgcn_wmma_f32_16x16x32_bf16(
            false, al[i].v, false, bh[j].v, (short)0, acc[i][j], false, false);
      }
    __syncthreads();
  }

  // epilogue: C/D 16x16 f32 layout => VGPR r: M = r + 8*half, N = lm
  #pragma unroll
  for (int i = 0; i < 2; ++i)
    #pragma unroll
    for (int j = 0; j < 4; ++j) {
      int gn = bn0 + wn + j * 16 + lm;
      float bb = (MODE == 1) ? bdec[gn] : 0.0f;
      int gm0 = bm0 + wm + i * 16 + half * 8;
      #pragma unroll
      for (int r = 0; r < 8; ++r)
        Cp[(size_t)(gm0 + r) * N + gn] = acc[i][j][r] + bb;
    }
}

// ---------------------------------------------------------------------------
// Exact global top-k threshold: 2-pass 16-bit radix histogram on f2key
// ---------------------------------------------------------------------------
__global__ void hist_hi_kernel(const float* __restrict__ xe,
                               unsigned* __restrict__ hist,
                               unsigned long long n) {
  unsigned long long i = (unsigned long long)blockIdx.x * blockDim.x + threadIdx.x;
  unsigned long long stride = (unsigned long long)gridDim.x * blockDim.x;
  for (; i < n; i += stride)
    atomicAdd(&hist[f2key(xe[i]) >> 16], 1u);
}

__global__ void hist_lo_kernel(const float* __restrict__ xe,
                               const unsigned* __restrict__ sel,
                               unsigned* __restrict__ hist,
                               unsigned long long n) {
  unsigned hi = sel[0];
  unsigned long long i = (unsigned long long)blockIdx.x * blockDim.x + threadIdx.x;
  unsigned long long stride = (unsigned long long)gridDim.x * blockDim.x;
  for (; i < n; i += stride) {
    unsigned key = f2key(xe[i]);
    if ((key >> 16) == hi) atomicAdd(&hist[key & 0xFFFFu], 1u);
  }
}

// Single-block descending scan of a 65536-bin histogram.
// pass 0: target = k*B  -> writes sel[0]=hi bin, sel[1]=remaining-in-bin
// pass 1: target = sel[1] -> writes sel[2]=lo bin, sel[4]=32-bit threshold key
__global__ void scan_kernel(const unsigned* __restrict__ hist,
                            unsigned* __restrict__ sel,
                            const int* __restrict__ kptr, int pass) {
  __shared__ unsigned csum[256];
  int t = threadIdx.x;
  unsigned long long target;
  if (pass == 0) {
    long long kk = kptr[0]; if (kk < 0) kk = 0;
    unsigned long long kt = (unsigned long long)kk * (unsigned long long)B_;
    unsigned long long n  = (unsigned long long)B_ * (unsigned long long)H_;
    if (kt > n) kt = n;
    if (kt == 0) kt = 1;
    target = kt;
  } else {
    target = sel[1];
    if (target == 0) target = 1;
  }
  unsigned s = 0;
  const unsigned* hp = hist + t * 256;
  for (int j = 0; j < 256; ++j) s += hp[j];
  csum[t] = s;
  __syncthreads();
  if (t == 0) {
    unsigned long long above = 0;
    int c = 255;
    while (c > 0 && above + csum[c] < target) { above += csum[c]; --c; }
    int b = c * 256 + 255;
    int lo = c * 256;
    while (b > lo && above + hist[b] < target) { above += hist[b]; --b; }
    if (pass == 0) {
      sel[0] = (unsigned)b;
      sel[1] = (unsigned)(target - above);
    } else {
      sel[2] = (unsigned)b;
      sel[4] = (sel[0] << 16) | (unsigned)b;   // final threshold key
    }
  }
}

// ---------------------------------------------------------------------------
extern "C" void kernel_launch(void* const* d_in, const int* in_sizes, int n_in,
                              void* d_out, int out_size, void* d_ws, size_t ws_size,
                              hipStream_t stream) {
  const float* x    = (const float*)d_in[0];   // (B, 2, D)   = (2048, 4096)
  const float* We   = (const float*)d_in[1];   // (2, D, H)   = (4096, 16384)
  const float* Wd   = (const float*)d_in[2];   // (H, 2, D)   = (16384, 4096)
  const float* benc = (const float*)d_in[3];   // (H,)
  const float* bdec = (const float*)d_in[4];   // (2, D) = 4096
  const int*   kp   = (const int*)d_in[5];     // scalar k on device
  float* out = (float*)d_out;                  // (B, 2, D)

  const size_t XE_BYTES = (size_t)B_ * (size_t)H_ * sizeof(float);  // 128 MB
  float*    xe  = (float*)d_ws;
  unsigned* h1  = (unsigned*)((char*)d_ws + XE_BYTES);
  unsigned* h2  = h1 + 65536;
  unsigned* sel = h2 + 65536;   // 8 u32

  // zero histograms + selection state each call (determinism)
  hipMemsetAsync(h1, 0, (2 * 65536 + 8) * sizeof(unsigned), stream);

  const unsigned long long NTOT = (unsigned long long)B_ * (unsigned long long)H_;

  // 1) encode: x_enc = x @ W_enc           (2048 x 16384, K = 4096)
  wmma_gemm_kernel<0><<<dim3(H_ / BN, B_ / BM), 256, 0, stream>>>(
      x, We, xe, B_, H_, MD_, nullptr, nullptr, nullptr);

  // 2) exact top-(k*B) threshold over x_enc
  hist_hi_kernel<<<2048, 256, 0, stream>>>(xe, h1, NTOT);
  scan_kernel<<<1, 256, 0, stream>>>(h1, sel, kp, 0);
  hist_lo_kernel<<<2048, 256, 0, stream>>>(xe, sel, h2, NTOT);
  scan_kernel<<<1, 256, 0, stream>>>(h2, sel, kp, 1);

  // 3) decode: out = (mask(x_enc)+b_enc) @ W_dec + b_dec   (2048 x 4096, K = 16384)
  wmma_gemm_kernel<1><<<dim3(MD_ / BN, B_ / BM), 256, 0, stream>>>(
      xe, Wd, out, B_, MD_, H_, benc, sel, bdec);
}
